// IntraRNN_40226663694649
// MI455X (gfx1250) — matrix operand order repaired and checked by reading, not compiled, not run on hardware
//
#include <hip/hip_runtime.h>
#include <hip/hip_bf16.h>

typedef __attribute__((ext_vector_type(16))) _Float16 v16h;
typedef __attribute__((ext_vector_type(8)))  float    v8f;

#define Hh   1024
#define H2   2048
#define H3   3072
#define Lh   8192
#define Vh   50000
#define Bh   2

// workspace layout (float offsets)
#define X_OFF     0        // 2 x 2048  rnn_in
#define GX_OFF    4096     // 2 x 3072
#define GH_OFF    10240    // 2 x 3072
#define HNEW_OFF  16384    // 2 x 1024
#define U_OFF     18432    // 1024      u = w_e^T v
#define E_OFF     19456    // 2 x 8192  energies -> attn weights (in place)
#define PART_OFF  35840    // 2 x 16 x 1024 context partials
#define CAT_OFF   68608    // 16 x 2048 _Float16 (= 16384 floats)

// d_out layout (float offsets): output, context, hidden, embedded, rnn_output
#define O_OUT     0
#define O_CTX     100000
#define O_HID     102048
#define O_EMB     104096
#define O_RNN     106144

union Frag { unsigned int u[8]; _Float16 h[16]; v16h v; };

// ---- K1: gather embedding + concat with last_context -> x ; write embedded out
__global__ void k_gather(const int* __restrict__ ids,
                         const float* __restrict__ emb,
                         const float* __restrict__ lctx,
                         float* __restrict__ ws, float* __restrict__ out) {
  int gid = blockIdx.x * 256 + threadIdx.x;          // 0..4095
  int b = gid >> 11, j = gid & 2047;
  float val;
  if (j < Hh) {
    val = emb[(size_t)ids[b] * Hh + j];
    out[O_EMB + b * Hh + j] = val;
  } else {
    val = lctx[b * Hh + (j - Hh)];
  }
  ws[X_OFF + gid] = val;
}

// ---- K2: gate GEMVs. warp per (m,b,j): m=0 -> gx = w_ih@x + b_ih (K=2048)
//                                        m=1 -> gh = w_hh@h + b_hh (K=1024)
__global__ void k_gates(const float* __restrict__ w_ih, const float* __restrict__ w_hh,
                        const float* __restrict__ b_ih, const float* __restrict__ b_hh,
                        const float* __restrict__ hprev, float* __restrict__ ws) {
  int gid  = blockIdx.x * 256 + threadIdx.x;
  int wid  = gid >> 5, lane = gid & 31;
  int m    = wid / (Bh * H3);
  int rem  = wid % (Bh * H3);
  int b    = rem / H3, j = rem % H3;
  const float* w;  const float* src;  int K;
  if (m == 0) { w = w_ih + (size_t)j * H2; src = ws + X_OFF + b * H2; K = H2; }
  else        { w = w_hh + (size_t)j * Hh; src = hprev + b * Hh;      K = Hh; }
  float s = 0.f;
  for (int k = lane; k < K; k += 32) s += w[k] * src[k];
  for (int off = 16; off > 0; off >>= 1) s += __shfl_down(s, off, 32);
  if (lane == 0) {
    if (m == 0) ws[GX_OFF + b * H3 + j] = s + b_ih[j];
    else        ws[GH_OFF + b * H3 + j] = s + b_hh[j];
  }
}

// ---- K3: GRU gate combine -> h_new ; write hidden + rnn_output
__global__ void k_gru(const float* __restrict__ hprev,
                      float* __restrict__ ws, float* __restrict__ out) {
  int gid = blockIdx.x * 256 + threadIdx.x;          // 0..2047
  int b = gid >> 10, i = gid & 1023;
  const float* gx = ws + GX_OFF + b * H3;
  const float* gh = ws + GH_OFF + b * H3;
  float r = 1.f / (1.f + expf(-(gx[i] + gh[i])));
  float z = 1.f / (1.f + expf(-(gx[Hh + i] + gh[Hh + i])));
  float n = tanhf(gx[2 * Hh + i] + r * gh[2 * Hh + i]);
  float hn = (1.f - z) * n + z * hprev[b * Hh + i];
  ws[HNEW_OFF + gid] = hn;
  out[O_HID + gid] = hn;
  out[O_RNN + gid] = hn;
}

// ---- K4: u[h] = sum_k attn_w[k, H+h] * v[k]   (softmax-invariant terms dropped)
__global__ void k_u(const float* __restrict__ attn_w, const float* __restrict__ v,
                    float* __restrict__ ws) {
  int h = blockIdx.x * 256 + threadIdx.x;            // 0..1023
  float s = 0.f;
  for (int k = 0; k < Hh; ++k) s += attn_w[(size_t)k * H2 + Hh + h] * v[k];
  ws[U_OFF + h] = s;
}

// ---- K5: energies[b,l] = dot(enc[b,l], u). warp per (b,l)
__global__ void k_energy(const float* __restrict__ enc, float* __restrict__ ws) {
  int gid = blockIdx.x * 256 + threadIdx.x;
  int wid = gid >> 5, lane = gid & 31;
  int b = wid >> 13, l = wid & (Lh - 1);
  const float* row = enc + ((size_t)b * Lh + l) * Hh;
  const float* u = ws + U_OFF;
  float s = 0.f;
  for (int k = lane; k < Hh; k += 32) s += row[k] * u[k];
  for (int off = 16; off > 0; off >>= 1) s += __shfl_down(s, off, 32);
  if (lane == 0) ws[E_OFF + b * Lh + l] = s;
}

// ---- K6: softmax over L, in place. one block per b
__global__ void k_softmax(float* __restrict__ ws) {
  int b = blockIdx.x, tid = threadIdx.x;
  float* row = ws + E_OFF + b * Lh;
  __shared__ float red[256];
  float m = -1e30f;
  for (int i = tid; i < Lh; i += 256) m = fmaxf(m, row[i]);
  red[tid] = m; __syncthreads();
  for (int s = 128; s > 0; s >>= 1) { if (tid < s) red[tid] = fmaxf(red[tid], red[tid + s]); __syncthreads(); }
  m = red[0]; __syncthreads();
  float sum = 0.f;
  for (int i = tid; i < Lh; i += 256) { float e = expf(row[i] - m); row[i] = e; sum += e; }
  red[tid] = sum; __syncthreads();
  for (int s = 128; s > 0; s >>= 1) { if (tid < s) red[tid] += red[tid + s]; __syncthreads(); }
  float inv = 1.f / red[0];
  for (int i = tid; i < Lh; i += 256) row[i] *= inv;
}

// ---- K7a: context partials: part[b,seg,h] = sum_{l in seg} w[b,l]*enc[b,l,h]
__global__ void k_ctx_part(const float* __restrict__ enc, float* __restrict__ ws) {
  int blk = blockIdx.x;                               // 0..127 = b*64 + seg*4 + hb
  int b = blk >> 6, rem = blk & 63;
  int seg = rem >> 2, hb = rem & 3;
  int h = hb * 256 + threadIdx.x;
  const float* aw = ws + E_OFF + b * Lh;
  float s = 0.f;
  int l0 = seg * (Lh / 16);
  for (int l = l0; l < l0 + (Lh / 16); ++l)
    s += aw[l] * enc[((size_t)b * Lh + l) * Hh + h];
  ws[PART_OFF + ((b * 16 + seg) * Hh) + h] = s;
}

// ---- K7b: reduce partials -> context (out) and build padded f16 cat (16 x 2048)
__global__ void k_cat(float* __restrict__ ws, float* __restrict__ out) {
  int gid = blockIdx.x * 256 + threadIdx.x;          // 0..32767
  int row = gid >> 11, j = gid & 2047;
  _Float16* cat = (_Float16*)(ws + CAT_OFF);
  float val = 0.f;
  if (row < Bh) {
    if (j < Hh) {
      val = ws[HNEW_OFF + row * Hh + j];
    } else {
      int h = j - Hh;
      float s = 0.f;
      for (int seg = 0; seg < 16; ++seg)
        s += ws[PART_OFF + ((row * 16 + seg) * Hh) + h];
      out[O_CTX + row * Hh + h] = s;
      val = s;
    }
  }
  cat[row * H2 + j] = (_Float16)val;
}

// ---- K8: output = cat16 @ out_w^T + out_b via v_wmma_f32_16x16x32_f16
// wave per 16-wide N tile (3125 tiles), K-loop 2048/32 = 64 WMMAs.
__global__ void k_outproj(const float* __restrict__ out_w, const float* __restrict__ out_b,
                          const float* __restrict__ ws, float* __restrict__ out) {
  int wave = (blockIdx.x * 256 + threadIdx.x) >> 5;
  int lane = threadIdx.x & 31;
  if (wave >= Vh / 16) return;                        // wave-uniform guard
  int g  = lane >> 4;                                 // half-wave group
  int nl = lane & 15;
  int n  = wave * 16 + nl;                            // output column (vocab id)
  int m  = nl;                                        // A row owned by this lane
  const unsigned int* catp = (const unsigned int*)(ws + CAT_OFF); // f16 pairs
  const float* wrow = out_w + (size_t)n * H2;
  v8f acc = {};
  for (int k0 = 0; k0 < H2; k0 += 32) {
    // A fragment: 16x32 f16, lane L holds row m=L&15; K pairs per ISA layout
    Frag a;
    const unsigned int* ap = catp + m * (H2 / 2) + (k0 >> 1) + g * 4;
    a.u[0] = ap[0]; a.u[1] = ap[1]; a.u[2] = ap[2];  a.u[3] = ap[3];
    a.u[4] = ap[8]; a.u[5] = ap[9]; a.u[6] = ap[10]; a.u[7] = ap[11];
    // B fragment: 32x16 f16, lane (g,nl) holds column n, K = k0+g*8+2v,+1 (v<4), k0+16+g*8+2(v-4),+1 (v>=4)
    Frag bm;
    const float* wp = wrow + k0 + g * 8;
#pragma unroll
    for (int vv = 0; vv < 8; ++vv) {
      float2 w2 = *(const float2*)(wp + 2 * vv + (vv >= 4 ? 8 : 0));
      bm.h[2 * vv]     = (_Float16)w2.x;
      bm.h[2 * vv + 1] = (_Float16)w2.y;
    }
    acc = __builtin_amdgcn_wmma_f32_16x16x32_f16(false, a.v, false, bm.v,
                                                 (short)0, acc, false, false);
  }
  if (g == 0) {                                       // lanes 0-15: VGPR r holds row M=r
    float bias = out_b[n];
    out[O_OUT + 0 * Vh + n] = acc[0] + bias;          // M=0 -> batch 0
    out[O_OUT + 1 * Vh + n] = acc[1] + bias;          // M=1 -> batch 1
  }
}

extern "C" void kernel_launch(void* const* d_in, const int* in_sizes, int n_in,
                              void* d_out, int out_size, void* d_ws, size_t ws_size,
                              hipStream_t stream) {
  const int*   ids    = (const int*)  d_in[0];
  const float* lctx   = (const float*)d_in[1];
  const float* lhid   = (const float*)d_in[2];
  const float* enc    = (const float*)d_in[3];
  const float* emb    = (const float*)d_in[4];
  const float* w_ih   = (const float*)d_in[5];
  const float* w_hh   = (const float*)d_in[6];
  const float* b_ih   = (const float*)d_in[7];
  const float* b_hh   = (const float*)d_in[8];
  const float* attn_w = (const float*)d_in[9];
  // d_in[10] = attn_b : per-row constant, cancels in softmax -> unused
  const float* v      = (const float*)d_in[11];
  const float* out_w  = (const float*)d_in[12];
  const float* out_b  = (const float*)d_in[13];
  float* out = (float*)d_out;
  float* ws  = (float*)d_ws;

  k_gather  <<<16,   256, 0, stream>>>(ids, emb, lctx, ws, out);
  k_gates   <<<1536, 256, 0, stream>>>(w_ih, w_hh, b_ih, b_hh, lhid, ws);
  k_gru     <<<8,    256, 0, stream>>>(lhid, ws, out);
  k_u       <<<4,    256, 0, stream>>>(attn_w, v, ws);
  k_energy  <<<2048, 256, 0, stream>>>(enc, ws);
  k_softmax <<<2,    256, 0, stream>>>(ws);
  k_ctx_part<<<128,  256, 0, stream>>>(enc, ws);
  k_cat     <<<128,  256, 0, stream>>>(ws, out);
  k_outproj <<<391,  256, 0, stream>>>(out_w, out_b, ws, out);
}
